// LRULayer_65687229825592
// MI455X (gfx1250) — matrix operand order repaired and checked by reading, compiled, not run on hardware
//
#include <hip/hip_runtime.h>
#include <cmath>

typedef __bf16 bf16;
typedef __attribute__((ext_vector_type(16))) __bf16 v16bf;
typedef __attribute__((ext_vector_type(8)))  __bf16 bf16x8;
typedef __attribute__((ext_vector_type(8)))  float   v8f;
typedef __attribute__((ext_vector_type(4)))  float   f32x4;

#define DM     2048      // d_model
#define DS     256       // d_state
#define BSZ    4
#define TL     2048
#define M_TOT  (BSZ*TL)  // 8192 rows
#define NCAT   (2*DS)    // 512 = [re | im] concatenated state
#define CHUNK  64        // scan chunk length
#define NCHUNK (TL/CHUNK) // 32

// ---------------------------------------------------------------------------
// Kernel 1: per-state parameters + power table A^(k+1) = r^(k+1) e^{i(k+1)θ}
// grid: (CHUNK) blocks x DS threads; block k computes power k+1
// ---------------------------------------------------------------------------
__global__ void lru_params(const float* __restrict__ nu_log,
                           const float* __restrict__ theta_log,
                           float* __restrict__ Are, float* __restrict__ Aim,
                           float* __restrict__ ALre, float* __restrict__ ALim,
                           float* __restrict__ Apr, float* __restrict__ Api) {
    int n = threadIdx.x;
    int k = blockIdx.x;                       // power = k+1
    float nu  = log1pf(expf(nu_log[n]));      // softplus
    float lnr = -expf(nu);                    // log|A|  (negative)
    float th  = expf(theta_log[n]);
    float p   = (float)(k + 1);
    float rp  = expf(lnr * p);
    Apr[k * DS + n] = rp * cosf(th * p);
    Api[k * DS + n] = rp * sinf(th * p);
    if (k == 0) {
        float mag = expf(lnr);
        Are[n] = mag * cosf(th);
        Aim[n] = mag * sinf(th);
        float rl = expf(lnr * (float)CHUNK);
        ALre[n] = rl * cosf(th * (float)CHUNK);
        ALim[n] = rl * sinf(th * (float)CHUNK);
    }
}

// ---------------------------------------------------------------------------
// Kernel 2: gB[n][d] = bf16(gamma[n]*B_re[n][d]); rows DS..2DS-1 = gamma*B_im
// grid: (DM/256, DS)
// ---------------------------------------------------------------------------
__global__ void pack_gB(const float* __restrict__ nu_log,
                        const float* __restrict__ B_re,
                        const float* __restrict__ B_im,
                        bf16* __restrict__ gB) {
    int n = blockIdx.y;
    int d = blockIdx.x * blockDim.x + threadIdx.x;
    float nu  = log1pf(expf(nu_log[n]));
    float mag = expf(-expf(nu));
    float g   = sqrtf(fmaxf(1.0f - mag * mag, 0.0f));
    size_t src = (size_t)n * DM + d;
    gB[src]                       = (bf16)(g * B_re[src]);
    gB[(size_t)(DS + n) * DM + d] = (bf16)(g * B_im[src]);
}

// ---------------------------------------------------------------------------
// Kernel 3: Wcat[d][j] = C_re[d][j] (j<DS) | -C_im[d][j-DS]   (bf16)
// ---------------------------------------------------------------------------
__global__ void pack_W(const float* __restrict__ C_re,
                       const float* __restrict__ C_im,
                       bf16* __restrict__ W) {
    int idx = blockIdx.x * 256 + threadIdx.x;     // over DM*NCAT
    int j = idx & (NCAT - 1);
    int d = idx >> 9;
    float v = (j < DS) ? C_re[(size_t)d * DS + j]
                       : -C_im[(size_t)d * DS + (j - DS)];
    W[idx] = (bf16)v;
}

// ---------------------------------------------------------------------------
// WMMA GEMM: C[M,N] = A[M,K] * B[N,K]^T   (f32 accumulate)
//   AF32: A operand is f32 in memory, converted to bf16 while staging to LDS
//   EPI : C[m][n] = acc + Dv[n] * xres[m*N+n]
// 256 threads = 8 waves (2x4); block tile 128x128; wave tile 64x32 (4x2 WMMA);
// K-step 32; double-buffered LDS, one barrier per K-step, register-staged
// global prefetch of the next K-step overlapping the 8 WMMAs.
// ---------------------------------------------------------------------------
template <bool EPI, bool AF32>
__global__ __launch_bounds__(256)
void gemm_wmma(const void* __restrict__ Aop, const bf16* __restrict__ Bm,
               float* __restrict__ C, int M, int N, int K,
               const float* __restrict__ xres, const float* __restrict__ Dv) {
    constexpr int TM = 128, TN = 128, TK = 32, LDA = 40; // 80B row stride: 16B aligned, conflict-free
    __shared__ alignas(16) bf16 As[2][TM * LDA];
    __shared__ alignas(16) bf16 Bs[2][TN * LDA];

    const bf16*  Ab = (const bf16*)Aop;
    const float* Af = (const float*)Aop;

    int tid  = threadIdx.x;
    int lane = tid & 31;
    int w    = tid >> 5;
    int wm0  = (w >> 2) * 64;     // 2 waves along M, wave tile M=64
    int wn0  = (w & 3) * 32;      // 4 waves along N, wave tile N=32
    int m0   = blockIdx.y * TM;
    int n0   = blockIdx.x * TN;
    int lrow  = lane & 15;
    int lhalf = lane >> 4;

    // loader mapping: each thread stages 16 elements of A and 16 of B
    int ldrow = tid >> 1, ldkc = (tid & 1) * 16;

    v8f acc[4][2] = {};
    f32x4  aregf[4];
    bf16x8 aregb[2];
    bf16x8 breg[2];

    auto preload = [&](int kb) {
        if constexpr (AF32) {
            const float* s = Af + (size_t)(m0 + ldrow) * K + kb + ldkc;
            aregf[0] = *(const f32x4*)(s);
            aregf[1] = *(const f32x4*)(s + 4);
            aregf[2] = *(const f32x4*)(s + 8);
            aregf[3] = *(const f32x4*)(s + 12);
        } else {
            const bf16* s = Ab + (size_t)(m0 + ldrow) * K + kb + ldkc;
            aregb[0] = *(const bf16x8*)(s);
            aregb[1] = *(const bf16x8*)(s + 8);
        }
        const bf16* sb = Bm + (size_t)(n0 + ldrow) * K + kb + ldkc;
        breg[0] = *(const bf16x8*)(sb);
        breg[1] = *(const bf16x8*)(sb + 8);
    };
    auto stage = [&](int buf) {
        bf16* pa = &As[buf][ldrow * LDA + ldkc];
        if constexpr (AF32) {
            bf16x8 o0, o1;
            #pragma unroll
            for (int i = 0; i < 4; ++i) {
                o0[i]     = (bf16)aregf[0][i];
                o0[4 + i] = (bf16)aregf[1][i];
                o1[i]     = (bf16)aregf[2][i];
                o1[4 + i] = (bf16)aregf[3][i];
            }
            *(bf16x8*)pa       = o0;
            *(bf16x8*)(pa + 8) = o1;
        } else {
            *(bf16x8*)pa       = aregb[0];
            *(bf16x8*)(pa + 8) = aregb[1];
        }
        bf16* pb = &Bs[buf][ldrow * LDA + ldkc];
        *(bf16x8*)pb       = breg[0];
        *(bf16x8*)(pb + 8) = breg[1];
    };

    preload(0);
    int buf = 0;
    for (int kb = 0; kb < K; kb += TK) {
        stage(buf);
        __syncthreads();
        if (kb + TK < K) preload(kb + TK);   // overlap next-tile loads with WMMAs

        v16bf afr[4], bfr[2];
        #pragma unroll
        for (int mt = 0; mt < 4; ++mt) {
            // A frag (16x32): lane half h holds K in {8h..8h+7} U {16+8h..16+8h+7}
            const bf16* p = &As[buf][(wm0 + mt * 16 + lrow) * LDA + lhalf * 8];
            ((bf16x8*)&afr[mt])[0] = *(const bf16x8*)p;
            ((bf16x8*)&afr[mt])[1] = *(const bf16x8*)(p + 16);
        }
        #pragma unroll
        for (int nt = 0; nt < 2; ++nt) {
            // B frag (32x16): lane half h holds K in {8h..8h+7} U {16+8h..16+8h+7} for column lane&15
            const bf16* p = &Bs[buf][(wn0 + nt * 16 + lrow) * LDA + lhalf * 8];
            ((bf16x8*)&bfr[nt])[0] = *(const bf16x8*)p;
            ((bf16x8*)&bfr[nt])[1] = *(const bf16x8*)(p + 16);
        }
        #pragma unroll
        for (int mt = 0; mt < 4; ++mt)
            #pragma unroll
            for (int nt = 0; nt < 2; ++nt)
                acc[mt][nt] = __builtin_amdgcn_wmma_f32_16x16x32_bf16(
                    false, afr[mt], false, bfr[nt],
                    (short)0, acc[mt][nt], false, false);
        buf ^= 1;
    }

    // C layout: VGPR r -> M = 8*(lane>=16)+r, N = lane&15
    #pragma unroll
    for (int mt = 0; mt < 4; ++mt)
        #pragma unroll
        for (int nt = 0; nt < 2; ++nt) {
            int row0 = m0 + wm0 + mt * 16 + lhalf * 8;
            int col  = n0 + wn0 + nt * 16 + lrow;
            #pragma unroll
            for (int r = 0; r < 8; ++r) {
                int row = row0 + r;
                float v = acc[mt][nt][r];
                if constexpr (EPI)
                    v += Dv[col] * xres[(size_t)row * N + col];
                C[(size_t)row * N + col] = v;
            }
        }
}

// ---------------------------------------------------------------------------
// Scan phase A: per (b, chunk, n) local scan of 64 steps, in place in Bu.
// grid: BSZ*NCHUNK blocks x DS threads
// ---------------------------------------------------------------------------
__global__ void scan_local(float* __restrict__ Bu, float* __restrict__ Fin,
                           const float* __restrict__ Are,
                           const float* __restrict__ Aim) {
    int n = threadIdx.x;
    int b = blockIdx.x >> 5, c = blockIdx.x & (NCHUNK - 1);
    float ar = Are[n], ai = Aim[n];
    float hr = 0.f, hi = 0.f;
    size_t base = ((size_t)(b * TL + c * CHUNK)) * NCAT + n;
    for (int t = 0; t < CHUNK; ++t) {
        float br = Bu[base], bi = Bu[base + DS];
        float nr = fmaf(ar, hr, fmaf(-ai, hi, br));
        float ni = fmaf(ar, hi, fmaf( ai, hr, bi));
        Bu[base] = nr; Bu[base + DS] = ni;
        hr = nr; hi = ni;
        base += NCAT;
    }
    size_t fo = (size_t)blockIdx.x * NCAT + n;
    Fin[fo] = hr; Fin[fo + DS] = hi;
}

// ---------------------------------------------------------------------------
// Scan phase B: carry[c] = A^CHUNK * carry[c-1] + Fin[c-1]  (32 serial steps)
// grid: BSZ blocks x DS threads
// ---------------------------------------------------------------------------
__global__ void scan_carry(const float* __restrict__ Fin, float* __restrict__ Carry,
                           const float* __restrict__ ALre,
                           const float* __restrict__ ALim) {
    int n = threadIdx.x, b = blockIdx.x;
    float alr = ALre[n], ali = ALim[n];
    float cr = 0.f, ci = 0.f;
    for (int c = 0; c < NCHUNK; ++c) {
        size_t o = (size_t)(b * NCHUNK + c) * NCAT + n;
        Carry[o] = cr; Carry[o + DS] = ci;
        float fr = Fin[o], fi = Fin[o + DS];
        float nr = alr * cr - ali * ci + fr;
        float ni = alr * ci + ali * cr + fi;
        cr = nr; ci = ni;
    }
}

// ---------------------------------------------------------------------------
// Scan phase C: H[t] += A^{k+1} * carry[chunk], emit bf16 Hcat for GEMM2.
// grid: M_TOT blocks x DS threads
// ---------------------------------------------------------------------------
__global__ void scan_fix(const float* __restrict__ H, const float* __restrict__ Carry,
                         const float* __restrict__ Apr, const float* __restrict__ Api,
                         bf16* __restrict__ Hbf) {
    int n = threadIdx.x;
    int m = blockIdx.x;
    int t = m & (TL - 1), b = m >> 11;
    int c = t >> 6, k = t & (CHUNK - 1);
    float pr = Apr[k * DS + n], pi = Api[k * DS + n];
    size_t co = (size_t)(b * NCHUNK + c) * NCAT + n;
    float cr = Carry[co], ci = Carry[co + DS];
    size_t ho = (size_t)m * NCAT + n;
    float hr = H[ho]      + (pr * cr - pi * ci);
    float hi = H[ho + DS] + (pr * ci + pi * cr);
    Hbf[ho]      = (bf16)hr;
    Hbf[ho + DS] = (bf16)hi;
}

// ---------------------------------------------------------------------------
extern "C" void kernel_launch(void* const* d_in, const int* in_sizes, int n_in,
                              void* d_out, int out_size, void* d_ws, size_t ws_size,
                              hipStream_t stream) {
    const float* x         = (const float*)d_in[0];
    const float* nu_log    = (const float*)d_in[1];
    const float* theta_log = (const float*)d_in[2];
    const float* B_re      = (const float*)d_in[3];
    const float* B_im      = (const float*)d_in[4];
    const float* C_re      = (const float*)d_in[5];
    const float* C_im      = (const float*)d_in[6];
    const float* Dvec      = (const float*)d_in[7];
    float* y = (float*)d_out;

    char* ws = (char*)d_ws;
    size_t off = 0;
    auto carve = [&](size_t bytes) -> void* {
        void* p = ws + off;
        off = (off + bytes + 255) & ~(size_t)255;
        return p;
    };
    bf16*  gB   = (bf16*) carve((size_t)NCAT * DM * sizeof(bf16));     //  2 MB
    bf16*  Wc   = (bf16*) carve((size_t)DM * NCAT * sizeof(bf16));     //  2 MB
    float* Bu   = (float*)carve((size_t)M_TOT * NCAT * sizeof(float)); // 16.8 MB (Bu -> H in place)
    bf16*  Hbf  = (bf16*) carve((size_t)M_TOT * NCAT * sizeof(bf16));  //  8.4 MB
    float* Are  = (float*)carve(DS * sizeof(float));
    float* Aim  = (float*)carve(DS * sizeof(float));
    float* ALre = (float*)carve(DS * sizeof(float));
    float* ALim = (float*)carve(DS * sizeof(float));
    float* Apr  = (float*)carve((size_t)CHUNK * DS * sizeof(float));
    float* Api  = (float*)carve((size_t)CHUNK * DS * sizeof(float));
    float* Fin  = (float*)carve((size_t)BSZ * NCHUNK * NCAT * sizeof(float));
    float* Carry= (float*)carve((size_t)BSZ * NCHUNK * NCAT * sizeof(float));

    // 1) parameters + power table
    lru_params<<<dim3(CHUNK), dim3(DS), 0, stream>>>(
        nu_log, theta_log, Are, Aim, ALre, ALim, Apr, Api);
    // 2) weight packing (f32 -> bf16)
    pack_gB<<<dim3(DM / 256, DS), dim3(256), 0, stream>>>(nu_log, B_re, B_im, gB);
    pack_W<<<dim3(DM * NCAT / 256), dim3(256), 0, stream>>>(C_re, C_im, Wc);
    // 3) input projection: Bu[M, 512] = x(f32->bf16) @ gB^T
    gemm_wmma<false, true><<<dim3(NCAT / 128, M_TOT / 128), dim3(256), 0, stream>>>(
        x, gB, Bu, M_TOT, NCAT, DM, nullptr, nullptr);
    // 4) blocked diagonal-complex scan (in place in Bu)
    scan_local<<<dim3(BSZ * NCHUNK), dim3(DS), 0, stream>>>(Bu, Fin, Are, Aim);
    scan_carry<<<dim3(BSZ), dim3(DS), 0, stream>>>(Fin, Carry, ALre, ALim);
    scan_fix<<<dim3(M_TOT), dim3(DS), 0, stream>>>(Bu, Carry, Apr, Api, Hbf);
    // 5) output projection: y = Hbf @ Wc^T + D*x
    gemm_wmma<true, false><<<dim3(DM / 128, M_TOT / 128), dim3(256), 0, stream>>>(
        Hbf, Wc, y, M_TOT, DM, NCAT, x, Dvec);
}
